// AutoEncoder_68521908240543
// MI455X (gfx1250) — compile-verified, matrix operand order checked by reading, not run on hardware
//
#include <hip/hip_runtime.h>
#include <hip/hip_bf16.h>

// ---------------------------------------------------------------------------
// VQ-VAE forward for MI455X (gfx1250, wave32, WMMA, TDM).
// GEMM-shaped work (VQ distances, 1x1 convs) -> v_wmma_f32_16x16x32_f16.
// VQ codebook staged into LDS via tensor_load_to_lds (TDM), waited with
// s_wait_tensorcnt; B-fragments then come from LDS (ds_load_b128).
// Spatial 4x4 convs -> VALU with LDS-staged weights.
// Conv biases feeding BatchNorm are dropped (they cancel exactly).
// ---------------------------------------------------------------------------

typedef __attribute__((ext_vector_type(16))) _Float16 v16h;
typedef __attribute__((ext_vector_type(8)))  float    v8f;
typedef __attribute__((ext_vector_type(4)))  unsigned v4u;
typedef __attribute__((ext_vector_type(4)))  int      v4i;
typedef __attribute__((ext_vector_type(8)))  int      v8i_t;

union AV { v16h v; unsigned u[8]; };

// 16-bit WMMA operand layout (ISA 7.12.2): lane -> M (or N) = lane&15,
// dword r holds k-pair base: r<4 -> 2r, r>=4 -> 16+2(r-4); lanes>=16 add +8.
__device__ inline void load_frag16(const _Float16* __restrict__ base, int hi, AV& av) {
    const int kofs = hi ? 8 : 0;
#pragma unroll
    for (int r = 0; r < 8; ++r) {
        int kb = ((r < 4) ? (2 * r) : (16 + 2 * (r - 4))) + kofs;
        av.u[r] = *reinterpret_cast<const unsigned*>(base + kb);
    }
}

// ------------------------------- small utils -------------------------------

__global__ void cvt_f16_kernel(const float* __restrict__ src, _Float16* __restrict__ dst, int n) {
    int i = blockIdx.x * blockDim.x + threadIdx.x;
    if (i < n) dst[i] = (_Float16)src[i];
}

__global__ void emb_sq_kernel(const float* __restrict__ emb, float* __restrict__ e2) {
    int n = blockIdx.x * blockDim.x + threadIdx.x;
    if (n < 512) {
        float s = 0.f;
#pragma unroll 8
        for (int k = 0; k < 64; ++k) { float v = emb[n * 64 + k]; s += v * v; }
        e2[n] = s;
    }
}

// ----------------------------- BatchNorm stats -----------------------------
// One block per channel; deterministic LDS tree reduction.
// stats[c] = mean, stats[64+c] = biased variance.
__global__ void bn_stats_kernel(const float* __restrict__ x, float* __restrict__ stats,
                                int C, int cnt, int S, int pixmajor) {
    int c = blockIdx.x;
    float s = 0.f, s2 = 0.f;
    for (int j = threadIdx.x; j < cnt; j += blockDim.x) {
        int off = pixmajor ? (j * C + c) : (((j / S) * C + c) * S + (j % S));
        float v = x[off];
        s += v; s2 += v * v;
    }
    __shared__ float sh[512];
    sh[threadIdx.x] = s; sh[256 + threadIdx.x] = s2;
    __syncthreads();
    for (int k = 128; k > 0; k >>= 1) {
        if ((int)threadIdx.x < k) {
            sh[threadIdx.x]       += sh[threadIdx.x + k];
            sh[256 + threadIdx.x] += sh[256 + threadIdx.x + k];
        }
        __syncthreads();
    }
    if (threadIdx.x == 0) {
        float m = sh[0] / (float)cnt;
        float v = sh[256] / (float)cnt - m * m;
        stats[c] = m; stats[64 + c] = v;
    }
}

// BN apply on NCHW, ReLU, f32 -> f32 (may be in-place).
__global__ void bn_apply_nchw_kernel(const float* __restrict__ src, float* __restrict__ dst,
                                     const float* __restrict__ stats, const float* __restrict__ g,
                                     const float* __restrict__ be, int C, int S, int total) {
    int i = blockIdx.x * blockDim.x + threadIdx.x;
    if (i >= total) return;
    int c = (i / S) % C;
    float v = g[c] * (src[i] - stats[c]) * rsqrtf(stats[64 + c] + 1e-5f) + be[c];
    dst[i] = fmaxf(v, 0.f);
}

// BN apply + ReLU, NCHW f32 -> pixel-major f16 (GEMM A staging).
__global__ void bn_nchw_to_pix_f16_kernel(const float* __restrict__ src, const float* __restrict__ stats,
                                          const float* __restrict__ g, const float* __restrict__ be,
                                          _Float16* __restrict__ dst, int C, int S, int total) {
    int i = blockIdx.x * blockDim.x + threadIdx.x;
    if (i >= total) return;
    int s = i % S; int c = (i / S) % C; int b = i / (S * C);
    float v = g[c] * (src[i] - stats[c]) * rsqrtf(stats[64 + c] + 1e-5f) + be[c];
    v = fmaxf(v, 0.f);
    dst[(b * S + s) * C + c] = (_Float16)v;
}

// BN apply on pixel-major input; optional ReLU; write NCHW f32 and/or pixel-major f16.
__global__ void bn_pix_apply_kernel(const float* __restrict__ src, const float* __restrict__ stats,
                                    const float* __restrict__ g, const float* __restrict__ be,
                                    float* __restrict__ out_nchw, _Float16* __restrict__ out_f16,
                                    int C, int S, int relu, int total) {
    int i = blockIdx.x * blockDim.x + threadIdx.x;
    if (i >= total) return;
    int c = i % C; int p = i / C;
    float v = g[c] * (src[i] - stats[c]) * rsqrtf(stats[64 + c] + 1e-5f) + be[c];
    if (relu) v = fmaxf(v, 0.f);
    if (out_f16)  out_f16[i] = (_Float16)v;
    if (out_nchw) { int b = p / S; int s = p % S; out_nchw[(b * C + c) * S + s] = v; }
}

// ------------------------------ spatial convs ------------------------------

// conv1: (32,1,256,256) -> (32,16,128,128), k=4 s=2 p=1, bias dropped (BN).
__global__ void conv1_kernel(const float* __restrict__ x, const float* __restrict__ w,
                             float* __restrict__ out) {
    int idx = blockIdx.x * blockDim.x + threadIdx.x;
    if (idx >= 32 * 16 * 128 * 128) return;
    int xo = idx & 127, yo = (idx >> 7) & 127, oc = (idx >> 14) & 15, b = idx >> 18;
    const float* xin = x + b * 65536;
    const float* wk  = w + oc * 16;
    float acc = 0.f;
#pragma unroll
    for (int ky = 0; ky < 4; ++ky) {
        int iy = yo * 2 - 1 + ky;
        if ((unsigned)iy >= 256u) continue;
#pragma unroll
        for (int kx = 0; kx < 4; ++kx) {
            int ix = xo * 2 - 1 + kx;
            if ((unsigned)ix >= 256u) continue;
            acc += xin[iy * 256 + ix] * wk[ky * 4 + kx];
        }
    }
    out[idx] = acc;
}

// conv2: (32,16,128,128) -> (32,32,64,64), k=4 s=2 p=1. Weights in LDS.
__global__ void conv2_kernel(const float* __restrict__ h1, const float* __restrict__ w,
                             float* __restrict__ out) {
    __shared__ float wl[8192];                 // 32*16*16 = 32 KB
    for (int i = threadIdx.x; i < 8192; i += blockDim.x) wl[i] = w[i];
    __syncthreads();
    int idx = blockIdx.x * blockDim.x + threadIdx.x;
    if (idx >= 32 * 32 * 64 * 64) return;
    int xo = idx & 63, yo = (idx >> 6) & 63, oc = (idx >> 12) & 31, b = idx >> 17;
    const float* hb = h1 + b * (16 * 16384);
    float acc = 0.f;
    for (int ic = 0; ic < 16; ++ic) {
        const float* plane = hb + ic * 16384;
        const float* wk = wl + (oc * 16 + ic) * 16;
#pragma unroll
        for (int ky = 0; ky < 4; ++ky) {
            int iy = yo * 2 - 1 + ky;
            if ((unsigned)iy >= 128u) continue;
#pragma unroll
            for (int kx = 0; kx < 4; ++kx) {
                int ix = xo * 2 - 1 + kx;
                if ((unsigned)ix >= 128u) continue;
                acc += plane[iy * 128 + ix] * wk[ky * 4 + kx];
            }
        }
    }
    out[idx] = acc;
}

// convT5: (32,32,64,64) -> (32,16,128,128), k=4 s=2 p=1, weight (Cin=32,Cout=16,4,4).
__global__ void convt5_kernel(const float* __restrict__ h4, const float* __restrict__ w5,
                              float* __restrict__ out) {
    __shared__ float wl[8192];                 // 32*16*16
    for (int i = threadIdx.x; i < 8192; i += blockDim.x) wl[i] = w5[i];
    __syncthreads();
    int idx = blockIdx.x * blockDim.x + threadIdx.x;
    if (idx >= 32 * 16 * 128 * 128) return;
    int xo = idx & 127, yo = (idx >> 7) & 127, oc = (idx >> 14) & 15, b = idx >> 18;
    int ky0 = (yo + 1) & 1, kx0 = (xo + 1) & 1;
    const float* hb = h4 + b * (32 * 4096);
    float acc = 0.f;
    for (int ic = 0; ic < 32; ++ic) {
        const float* plane = hb + ic * 4096;
        const float* wk = wl + (ic * 16 + oc) * 16;
#pragma unroll
        for (int dy = 0; dy < 2; ++dy) {
            int ky = ky0 + 2 * dy;
            int h = (yo + 1 - ky) >> 1;
            if ((unsigned)h >= 64u) continue;
#pragma unroll
            for (int dx = 0; dx < 2; ++dx) {
                int kx = kx0 + 2 * dx;
                int wv = (xo + 1 - kx) >> 1;
                if ((unsigned)wv >= 64u) continue;
                acc += plane[h * 64 + wv] * wk[ky * 4 + kx];
            }
        }
    }
    out[idx] = acc;
}

// convT6: (32,16,128,128) -> (32,1,256,256), + b6, sigmoid.
__global__ void convt6_kernel(const float* __restrict__ h5, const float* __restrict__ w6,
                              const float* __restrict__ b6, float* __restrict__ out) {
    int idx = blockIdx.x * blockDim.x + threadIdx.x;
    if (idx >= 32 * 256 * 256) return;
    int xo = idx & 255, yo = (idx >> 8) & 255, b = idx >> 16;
    int ky0 = (yo + 1) & 1, kx0 = (xo + 1) & 1;
    const float* hb = h5 + b * (16 * 16384);
    float acc = b6[0];
    for (int ic = 0; ic < 16; ++ic) {
        const float* plane = hb + ic * 16384;
        const float* wk = w6 + ic * 16;
#pragma unroll
        for (int dy = 0; dy < 2; ++dy) {
            int ky = ky0 + 2 * dy;
            int h = (yo + 1 - ky) >> 1;
            if ((unsigned)h >= 128u) continue;
#pragma unroll
            for (int dx = 0; dx < 2; ++dx) {
                int kx = kx0 + 2 * dx;
                int wv = (xo + 1 - kx) >> 1;
                if ((unsigned)wv >= 128u) continue;
                acc += plane[h * 128 + wv] * wk[ky * 4 + kx];
            }
        }
    }
    out[idx] = 1.0f / (1.0f + __expf(-acc));
}

// ------------------------------- WMMA GEMMs --------------------------------
// C[M][N] = A[M][K] x B[N][K]^T.  One wave per 16x16 tile; K in {32,64}.
__global__ __launch_bounds__(32) void gemm_f16_wmma_kernel(
    const _Float16* __restrict__ A, const _Float16* __restrict__ B,
    float* __restrict__ C, int N, int K) {
    int mt = blockIdx.x, nt = blockIdx.y;
    int lane = threadIdx.x, hi = lane >> 4, col = lane & 15;
    const _Float16* arow = A + (mt * 16 + col) * K;
    const _Float16* brow = B + (nt * 16 + col) * K;
    v8f acc = {};
    for (int k0 = 0; k0 < K; k0 += 32) {
        AV a, b;
        load_frag16(arow + k0, hi, a);
        load_frag16(brow + k0, hi, b);
        acc = __builtin_amdgcn_wmma_f32_16x16x32_f16(false, a.v, false, b.v,
                                                     (short)0, acc, false, false);
    }
#pragma unroll
    for (int r = 0; r < 8; ++r)
        C[(mt * 16 + r + 8 * hi) * N + nt * 16 + col] = acc[r];
}

// Fused VQ: codebook (512x64 f16, 64 KB) DMA'd to LDS by the Tensor Data
// Mover once per workgroup; 8 waves then sweep all 512 codes with WMMA,
// dist = e2[n] - 2*cross (||z||^2 drops out of argmin), running argmin in
// registers, cross-lane min via shuffles.
__global__ __launch_bounds__(256) void vq_argmin_kernel(
    const _Float16* __restrict__ zeF16,   // [P][64]
    const _Float16* __restrict__ embF16,  // [512][64]
    const float* __restrict__ e2,         // [512]
    int* __restrict__ latents, int P) {
    __shared__ __align__(16) _Float16 semb[512 * 64];   // 64 KB codebook
    __shared__ float se2[512];

    int tid = threadIdx.x;
#if __has_builtin(__builtin_amdgcn_tensor_load_to_lds) && __has_builtin(__builtin_amdgcn_s_wait_tensorcnt)
    if (tid < 32) {                      // wave 0 issues one TDM descriptor
        unsigned lds_base = (unsigned)(unsigned long long)&semb[0];
        unsigned long long ga = (unsigned long long)embF16;
        v4u g0;
        g0[0] = 1u;                                   // count=1 (valid D#)
        g0[1] = lds_base;                             // lds_addr
        g0[2] = (unsigned)ga;                         // global_addr[31:0]
        g0[3] = (unsigned)((ga >> 32) & 0x01FFFFFFu)  // global_addr[56:32]
                | 0x80000000u;                        // type=2 ("image")
        v8i_t g1;
        g1[0] = 0x00010000;                           // data_size=1 -> 2 bytes
        g1[1] = (int)(64u << 16);                     // tensor_dim0 = 64
        g1[2] = (int)(512u << 16);                    // tensor_dim1 = 512
        g1[3] = (int)(64u << 16);                     // tile_dim0 = 64
        g1[4] = 512;                                  // tile_dim1 = 512
        g1[5] = 64;                                   // tensor_dim0_stride = 64
        g1[6] = 0;
        g1[7] = 0;
        v4i  gz4 = {0, 0, 0, 0};                      // 2-D tile: groups 2/3 unused
        v8i_t gz8 = {0, 0, 0, 0, 0, 0, 0, 0};         // extra group (6-arg builtin)
        __builtin_amdgcn_tensor_load_to_lds(g0, g1, gz4, gz4, gz8, 0);
        __builtin_amdgcn_s_wait_tensorcnt((short)0);  // s_wait_tensorcnt 0
    }
#else
    {
        const uint4* s = (const uint4*)embF16;
        uint4* d = (uint4*)semb;
        for (int i = tid; i < 4096; i += 256) d[i] = s[i];
    }
#endif
    for (int i = tid; i < 512; i += 256) se2[i] = e2[i];
    __syncthreads();

    int wave = tid >> 5;
    int lane = tid & 31, hi = lane >> 4, col = lane & 15;
    int tile = blockIdx.x * 8 + wave;
    int rowbase = tile * 16;
    const _Float16* arow = zeF16 + (rowbase + col) * 64;
    __builtin_prefetch(arow, 0, 1);                   // global_prefetch_b8
    AV a0, a1;
    load_frag16(arow + 0,  hi, a0);
    load_frag16(arow + 32, hi, a1);
    float bestv[8]; int besti[8];
#pragma unroll
    for (int r = 0; r < 8; ++r) { bestv[r] = 3.4e38f; besti[r] = 0; }
    for (int n0 = 0; n0 < 512; n0 += 16) {
        const _Float16* brow = &semb[(n0 + col) * 64];  // LDS: ds_load_b128
        AV b0, b1;
        load_frag16(brow + 0,  hi, b0);
        load_frag16(brow + 32, hi, b1);
        v8f acc = {};
        acc = __builtin_amdgcn_wmma_f32_16x16x32_f16(false, a0.v, false, b0.v,
                                                     (short)0, acc, false, false);
        acc = __builtin_amdgcn_wmma_f32_16x16x32_f16(false, a1.v, false, b1.v,
                                                     (short)0, acc, false, false);
        float e2n = se2[n0 + col];
#pragma unroll
        for (int r = 0; r < 8; ++r) {
            float d = e2n - 2.0f * acc[r];
            int   idx = n0 + col;
#pragma unroll
            for (int m = 1; m < 16; m <<= 1) {   // min over the 16-lane half
                float dv = __shfl_xor(d, m, 32);
                int   di = __shfl_xor(idx, m, 32);
                if (dv < d || (dv == d && di < idx)) { d = dv; idx = di; }
            }
            if (d < bestv[r]) { bestv[r] = d; besti[r] = idx; }
        }
    }
    if (col == 0) {
#pragma unroll
        for (int r = 0; r < 8; ++r)
            latents[rowbase + r + 8 * hi] = besti[r];
    }
}

// z_q gather: NCHW f32 output + pixel-major f16 A-matrix for conv4.
__global__ void zq_gather_kernel(const int* __restrict__ latents, const float* __restrict__ emb,
                                 float* __restrict__ zq_nchw, _Float16* __restrict__ A4, int P) {
    int i = blockIdx.x * blockDim.x + threadIdx.x;
    if (i >= P * 64) return;
    int c = i & 63, p = i >> 6;
    float v = emb[latents[p] * 64 + c];
    A4[i] = (_Float16)v;
    int b = p >> 12, s = p & 4095;
    zq_nchw[(b * 64 + c) * 4096 + s] = v;
}

// ---------------------------------------------------------------------------

extern "C" void kernel_launch(void* const* d_in, const int* in_sizes, int n_in,
                              void* d_out, int out_size, void* d_ws, size_t ws_size,
                              hipStream_t stream) {
    const float* x   = (const float*)d_in[0];
    const float* w1  = (const float*)d_in[1];
    const float* g1  = (const float*)d_in[3];
    const float* be1 = (const float*)d_in[4];
    const float* w2  = (const float*)d_in[5];
    const float* g2  = (const float*)d_in[7];
    const float* be2 = (const float*)d_in[8];
    const float* w3  = (const float*)d_in[9];
    const float* g3  = (const float*)d_in[11];
    const float* be3 = (const float*)d_in[12];
    const float* emb = (const float*)d_in[13];
    const float* w4  = (const float*)d_in[14];
    const float* g4  = (const float*)d_in[16];
    const float* be4 = (const float*)d_in[17];
    const float* w5  = (const float*)d_in[18];
    const float* g5  = (const float*)d_in[20];
    const float* be5 = (const float*)d_in[21];
    const float* w6  = (const float*)d_in[22];
    const float* b6  = (const float*)d_in[23];

    float* out_xt = (float*)d_out;            // (32,1,256,256)
    float* out_ze = out_xt + 2097152;         // (32,64,64,64)
    float* out_zq = out_ze + 8388608;         // (32,64,64,64)

    char* ws = (char*)d_ws;
    // Overlaid scratch (lifetimes are disjoint):
    float*    h1     = (float*)(ws + 0);                  // 33.5 MB  (later A4)
    _Float16* A4     = (_Float16*)(ws + 0);               //   reuse of h1
    float*    h2raw  = (float*)(ws + 33554432);           // 16.8 MB  (later h4 NCHW)
    float*    h4nchw = (float*)(ws + 33554432);           //   reuse
    _Float16* A2     = (_Float16*)(ws + 50331648);        // 8.4 MB   (later h4raw)
    float*    h4raw  = (float*)(ws + 50331648);           //   reuse (16.8 MB slot)
    float*    zraw   = (float*)(ws + 67108864);           // 33.5 MB  (later h5)
    float*    h5     = (float*)(ws + 67108864);           //   reuse
    _Float16* zeF16  = (_Float16*)(ws + 100663296);       // 16.8 MB
    _Float16* embF16 = (_Float16*)(ws + 117440512);       // 64 KB
    float*    e2     = (float*)(ws + 117506048);          // 2 KB
    _Float16* W3f16  = (_Float16*)(ws + 117508096);       // 4 KB
    _Float16* W4f16  = (_Float16*)(ws + 117512192);       // 4 KB
    int*      lat    = (int*)(ws + 117516288);            // 512 KB
    float*    stats  = (float*)(ws + 118040576);          // 512 B

    const int P = 131072;  // B * 64 * 64 pixels at latent resolution

    // f16 staging of codebook + GEMM weights, ||e||^2
    cvt_f16_kernel<<<128, 256, 0, stream>>>(emb, embF16, 32768);
    cvt_f16_kernel<<<8, 256, 0, stream>>>(w3, W3f16, 2048);
    cvt_f16_kernel<<<8, 256, 0, stream>>>(w4, W4f16, 2048);
    emb_sq_kernel<<<2, 256, 0, stream>>>(emb, e2);

    // ---- encoder ----
    conv1_kernel<<<32768, 256, 0, stream>>>(x, w1, h1);
    bn_stats_kernel<<<16, 256, 0, stream>>>(h1, stats, 16, 524288, 16384, 0);
    bn_apply_nchw_kernel<<<32768, 256, 0, stream>>>(h1, h1, stats, g1, be1, 16, 16384, 8388608);

    conv2_kernel<<<16384, 256, 0, stream>>>(h1, w2, h2raw);
    bn_stats_kernel<<<32, 256, 0, stream>>>(h2raw, stats, 32, 131072, 4096, 0);
    bn_nchw_to_pix_f16_kernel<<<16384, 256, 0, stream>>>(h2raw, stats, g2, be2, A2, 32, 4096, 4194304);

    // conv3 (1x1) as WMMA GEMM: M=P, N=64, K=32
    gemm_f16_wmma_kernel<<<dim3(P / 16, 4), 32, 0, stream>>>(A2, W3f16, zraw, 64, 32);
    bn_stats_kernel<<<64, 256, 0, stream>>>(zraw, stats, 64, P, 0, 1);
    bn_pix_apply_kernel<<<32768, 256, 0, stream>>>(zraw, stats, g3, be3, out_ze, zeF16, 64, 4096, 0, 8388608);

    // ---- vector quantization (TDM -> LDS codebook, WMMA + fused argmin) ----
    vq_argmin_kernel<<<P / 128, 256, 0, stream>>>(zeF16, embF16, e2, lat, P);
    zq_gather_kernel<<<32768, 256, 0, stream>>>(lat, emb, out_zq, A4, P);

    // ---- decoder ----
    // conv4 (1x1) as WMMA GEMM: M=P, N=32, K=64
    gemm_f16_wmma_kernel<<<dim3(P / 16, 2), 32, 0, stream>>>(A4, W4f16, h4raw, 32, 64);
    bn_stats_kernel<<<32, 256, 0, stream>>>(h4raw, stats, 32, P, 0, 1);
    bn_pix_apply_kernel<<<16384, 256, 0, stream>>>(h4raw, stats, g4, be4, h4nchw, ((_Float16*)0), 32, 4096, 1, 4194304);

    convt5_kernel<<<32768, 256, 0, stream>>>(h4nchw, w5, h5);
    bn_stats_kernel<<<16, 256, 0, stream>>>(h5, stats, 16, 524288, 16384, 0);
    bn_apply_nchw_kernel<<<32768, 256, 0, stream>>>(h5, h5, stats, g5, be5, 16, 16384, 8388608);

    convt6_kernel<<<8192, 256, 0, stream>>>(h5, w6, b6, out_xt);

    (void)in_sizes; (void)n_in; (void)out_size; (void)ws_size;
}
